// LPN3D_62457414418642
// MI455X (gfx1250) — matrix operand order repaired and checked by reading, not compiled, not run on hardware
//
#include <hip/hip_runtime.h>
#include <hip/hip_bf16.h>
#include <math.h>

typedef __attribute__((ext_vector_type(16))) _Float16 v16h;
typedef __attribute__((ext_vector_type(8)))  _Float16 v8h;
typedef __attribute__((ext_vector_type(8)))  float    v8f;
typedef unsigned int u32x4 __attribute__((ext_vector_type(4)));
typedef int          i32x8 __attribute__((ext_vector_type(8)));
typedef int          i32x4 __attribute__((ext_vector_type(4)));

#define DD   16
#define HH   64
#define WW   64
#define CIN  256
#define NVOX (DD*HH*WW)           // 65536
#define LROW 40                   // 32 data + 8 pad halfs per LDS row (64B + 16B)
#define NITER 216                 // 27 taps * 8 k-blocks
#define PRE_NMS_TOPK 2048
#define MAX_OUTPUT   256

#if __has_builtin(__builtin_amdgcn_tensor_load_to_lds)
#define HAVE_TDM 1
#else
#define HAVE_TDM 0
#endif

__device__ __forceinline__ float gelu_tanh(float x) {
  float x3 = x * x * x;
  return 0.5f * x * (1.f + tanhf(0.7978845608028654f * (x + 0.044715f * x3)));
}

#if HAVE_TDM
// DMA one contiguous 8192-half (16KB) weight panel into LDS, inserting a 16B pad
// after every 64B so the LDS image matches the [256][LROW] padded row layout.
__device__ __forceinline__ void tdm_load_panel(const _Float16* gsrc, unsigned lds_off) {
  unsigned long long ga = (unsigned long long)(size_t)gsrc;
  u32x4 g0;
  g0[0] = 1u;                                              // count=1, user descriptor
  g0[1] = lds_off;                                         // LDS byte address
  g0[2] = (unsigned)(ga & 0xFFFFFFFFull);                  // global_addr[31:0]
  g0[3] = (unsigned)((ga >> 32) & 0x1FFFFFFull) | (2u << 30);  // addr[56:32] | type=2
  i32x8 g1;
  g1[0] = (int)((1u << 16)      // data_size = 2 bytes
              | (1u << 20)      // pad_enable
              | (3u << 22)      // pad_interval: 16 DWORDs (64B) between pads
              | (3u << 25));    // pad_amount: 4 DWORDs (16B)
  g1[1] = (int)(0x2000u << 16); // tensor_dim0 = 8192 (lo16 in [31:16])
  g1[2] = (int)(1u << 16);      // tensor_dim0 hi16 = 0 | tensor_dim1 = 1
  g1[3] = (int)(0x2000u << 16); // tensor_dim1 hi16 = 0 | tile_dim0 = 8192
  g1[4] = 1;                    // tile_dim1 = 1 | tile_dim2 = 0
  g1[5] = 8192;                 // tensor_dim0_stride lo32
  g1[6] = 0;                    // stride hi16 | tensor_dim1_stride lo16
  g1[7] = 0;
  i32x4 z4 = {0, 0, 0, 0};      // groups 2/3 unused (<=2D tensor)
#if __clang_major__ >= 23
  i32x8 z8 = {0, 0, 0, 0, 0, 0, 0, 0};
  __builtin_amdgcn_tensor_load_to_lds(g0, g1, z4, z4, z8, 0);
#else
  __builtin_amdgcn_tensor_load_to_lds(g0, g1, z4, z4, 0);
#endif
}
#endif

// ---------------------------------------------------------------- conversions
__global__ __launch_bounds__(256) void cvt_f32_f16(const float* __restrict__ in,
                                                   _Float16* __restrict__ out, int n) {
  int i = blockIdx.x * 256 + threadIdx.x;
  if (i < n) out[i] = (_Float16)in[i];
}

// w [27][cin=256][cout=256] f32 -> wr [27][cin/32][cout=256][32] f16
__global__ __launch_bounds__(256) void reformat_w(const float* __restrict__ w,
                                                  _Float16* __restrict__ out) {
  int i = blockIdx.x * 256 + threadIdx.x;          // 27*256*256 total
  if (i >= 27 * 256 * 256) return;
  int co = i & 255, ci = (i >> 8) & 255, t = i >> 16;
  float v = w[(size_t)(t * 256 + ci) * 256 + co];
  out[((size_t)(t * 8 + (ci >> 5)) * 256 + co) * 32 + (ci & 31)] = (_Float16)v;
}

// ---------------------------------------------------------------- conv3d (implicit GEMM, WMMA + TDM)
__global__ __launch_bounds__(256) void conv3_wmma(const _Float16* __restrict__ x,
                                                  const _Float16* __restrict__ wr,
                                                  const float* __restrict__ bias,
                                                  _Float16* __restrict__ y) {
  __shared__ __align__(16) _Float16 lA[64][LROW];        // 64 voxels x 32 cin
  __shared__ __align__(16) _Float16 lB[2][256][LROW];    // double-buffered weight panel

  const int tid  = threadIdx.x;
  const int d    = blockIdx.x / HH;
  const int h    = blockIdx.x % HH;

  const int lane = tid & 31;
  const int wv   = tid >> 5;
  const int wvM  = wv >> 2;                          // 0..1  (32 voxels)
  const int wvN  = wv & 3;                           // 0..3  (64 couts)
  const int lrow = lane & 15;
  const int lhi  = lane >> 4;

  v8f acc[2][4];
  for (int a = 0; a < 2; ++a)
    for (int b = 0; b < 4; ++b) acc[a][b] = (v8f)0.f;

  const int arow = tid >> 2;                         // 0..63 voxel in row
  const int aseg = tid & 3;                          // 16B segment of 32 cin

#if HAVE_TDM
  if (wv == 0)                                       // prologue: DMA panel 0
    tdm_load_panel(wr, (unsigned)(size_t)&lB[0][0][0]);
#endif

  for (int it = 0; it < NITER; ++it) {
    const int t  = it >> 3, kb = it & 7;
    const int dz = t / 9 - 1, dy = (t / 3) % 3 - 1, dx = t % 3 - 1;
    const int d2 = d + dz, h2 = h + dy, w2 = arow + dx;
    const bool inb = ((unsigned)d2 < DD) && ((unsigned)h2 < HH) && ((unsigned)w2 < WW);

    __syncthreads();   // previous compute done: safe to overwrite lA / next B buffer

    // stage A tile (boundary-masked, per-lane)
    uint4 av = {0u, 0u, 0u, 0u};
    if (inb) av = *(const uint4*)(x + ((size_t)(d2 * HH + h2) * WW + w2) * CIN +
                                  kb * 32 + aseg * 8);
    *(uint4*)&lA[arow][aseg * 8] = av;

#if HAVE_TDM
    if (wv == 0) {
      if (it + 1 < NITER) {                          // DMA next panel into other buffer
        tdm_load_panel(wr + (size_t)(it + 1) * 8192,
                       (unsigned)(size_t)&lB[(it + 1) & 1][0][0]);
        __builtin_amdgcn_s_wait_tensorcnt(1);        // current panel landed
      } else {
        __builtin_amdgcn_s_wait_tensorcnt(0);
      }
    }
#else
    {
      const uint4* wp = (const uint4*)(wr + ((size_t)it * 256 + tid) * 32);
      _Float16* dst = &lB[it & 1][tid][0];
      ((uint4*)dst)[0] = wp[0];
      *(uint4*)(dst + 8)  = wp[1];
      *(uint4*)(dst + 16) = wp[2];
      *(uint4*)(dst + 24) = wp[3];
    }
#endif
    __syncthreads();   // lA stores + current B panel visible

    const _Float16 (*lBc)[LROW] = lB[it & 1];

    // A fragments: lanes<16 hold K0-7/16-23, lanes>=16 hold K8-15/24-31
    v16h afr[2], bfr[4];
    const int ko = lhi * 8;
    for (int mt = 0; mt < 2; ++mt) {
      const int r = wvM * 32 + mt * 16 + lrow;
      v8h lo = *(const v8h*)&lA[r][ko];
      v8h hi = *(const v8h*)&lA[r][ko + 16];
      for (int e = 0; e < 8; ++e) { afr[mt][e] = lo[e]; afr[mt][e + 8] = hi[e]; }
    }
    // B fragments: lanes<16 hold K0-7/16-23, lanes>=16 hold K8-15/24-31, N = lrow
    const int kb2 = lhi * 8;
    for (int nt = 0; nt < 4; ++nt) {
      const int n = wvN * 64 + nt * 16 + lrow;
      v8h lo = *(const v8h*)&lBc[n][kb2];
      v8h hi = *(const v8h*)&lBc[n][kb2 + 16];
      for (int e = 0; e < 8; ++e) { bfr[nt][e] = lo[e]; bfr[nt][e + 8] = hi[e]; }
    }
    for (int mt = 0; mt < 2; ++mt)
      for (int nt = 0; nt < 4; ++nt)
        acc[mt][nt] = __builtin_amdgcn_wmma_f32_16x16x32_f16(
            false, afr[mt], false, bfr[nt], (short)0, acc[mt][nt], false, false);
  }

  // epilogue: bias + gelu, store f16 (coalesced across 16 lanes)
  const size_t obase = (size_t)(d * HH + h) * WW;
  for (int mt = 0; mt < 2; ++mt)
    for (int nt = 0; nt < 4; ++nt) {
      const int n = wvN * 64 + nt * 16 + lrow;
      const float bn = bias[n];
      for (int r = 0; r < 8; ++r) {
        const int m = wvM * 32 + mt * 16 + r + lhi * 8;
        y[(obase + m) * CIN + n] = (_Float16)gelu_tanh(acc[mt][nt][r] + bn);
      }
    }
}

// ---------------------------------------------------------------- detection head
__global__ __launch_bounds__(256) void head_kernel(const _Float16* __restrict__ x,
                                                   const float* __restrict__ wl,
                                                   const float* __restrict__ bl,
                                                   const float* __restrict__ wr3,
                                                   const float* __restrict__ br,
                                                   float* __restrict__ scores,
                                                   float* __restrict__ locs) {
  const int v = blockIdx.x * 256 + threadIdx.x;     // one voxel per thread
  const v8h* xp = (const v8h*)(x + (size_t)v * CIN);
  float al = 0.f, a0 = 0.f, a1 = 0.f, a2 = 0.f;
  for (int c8 = 0; c8 < 32; ++c8) {
    v8h xv = xp[c8];
    for (int e = 0; e < 8; ++e) {
      const int c = c8 * 8 + e;
      const float f = (float)xv[e];
      al += f * wl[c];
      a0 += f * wr3[c * 3 + 0];
      a1 += f * wr3[c * 3 + 1];
      a2 += f * wr3[c * 3 + 2];
    }
  }
  al += bl[0]; a0 += br[0]; a1 += br[1]; a2 += br[2];
  scores[v] = 1.f / (1.f + expf(-al));
  const int wx = v & 63, hy = (v >> 6) & 63, dz = v >> 12;
  locs[v * 3 + 0] = (dz + 0.5f) * 4.f + a0 * 4.f;
  locs[v * 3 + 1] = (hy + 0.5f) * 4.f + a1 * 4.f;
  locs[v * 3 + 2] = (wx + 0.5f) * 4.f + a2 * 4.f;
}

// ---------------------------------------------------------------- top-k selection
__global__ __launch_bounds__(256) void init_sel(int* __restrict__ hist, int* __restrict__ ctr,
                                                float* __restrict__ cs, int* __restrict__ cidx) {
  int i = blockIdx.x * 256 + threadIdx.x;           // 8 blocks = 2048 threads
  if (i < 1024) hist[i] = 0;
  if (i < 8)    ctr[i]  = 0;
  if (i < PRE_NMS_TOPK) { cs[i] = 0.f; cidx[i] = 0; }
}

__global__ __launch_bounds__(256) void hist_kernel(const float* __restrict__ scores,
                                                   int* __restrict__ hist) {
  int v = blockIdx.x * 256 + threadIdx.x;
  unsigned bits = __float_as_uint(scores[v]);
  int bin = (int)(bits >> 20); if (bin > 1023) bin = 1023;
  atomicAdd(&hist[bin], 1);
}

__global__ void scan_kernel(const int* __restrict__ hist, int* __restrict__ ctr) {
  if (threadIdx.x == 0) {
    int cum = 0, t = 0, cAbove = 0;
    for (int b = 1023; b >= 0; --b) {
      cum += hist[b];
      if (cum >= PRE_NMS_TOPK) { t = b; cAbove = cum - hist[b]; break; }
    }
    ctr[2] = t; ctr[3] = cAbove; ctr[4] = PRE_NMS_TOPK - cAbove;
    ctr[0] = 0; ctr[1] = 0;
  }
}

__global__ __launch_bounds__(256) void compact_kernel(const float* __restrict__ scores,
                                                      int* __restrict__ ctr,
                                                      float* __restrict__ cs,
                                                      int* __restrict__ cidx) {
  int v = blockIdx.x * 256 + threadIdx.x;
  float s = scores[v];
  unsigned bits = __float_as_uint(s);
  int bin = (int)(bits >> 20); if (bin > 1023) bin = 1023;
  const int t = ctr[2], cAbove = ctr[3], needEq = ctr[4];
  if (bin > t) {
    int p = atomicAdd(&ctr[0], 1);
    if (p < PRE_NMS_TOPK) { cs[p] = s; cidx[p] = v; }
  } else if (bin == t) {
    int q = atomicAdd(&ctr[1], 1);
    if (q < needEq) { cs[cAbove + q] = s; cidx[cAbove + q] = v; }
  }
}

// bitonic sort 2048 entries descending (one WG, 1024 threads)
__global__ __launch_bounds__(1024) void sort_kernel(float* __restrict__ cs,
                                                    int* __restrict__ cidx) {
  __shared__ float ss[PRE_NMS_TOPK];
  __shared__ int   si[PRE_NMS_TOPK];
  const int t = threadIdx.x;
  ss[t] = cs[t]; ss[t + 1024] = cs[t + 1024];
  si[t] = cidx[t]; si[t + 1024] = cidx[t + 1024];
  for (int size = 2; size <= PRE_NMS_TOPK; size <<= 1)
    for (int stride = size >> 1; stride > 0; stride >>= 1) {
      __syncthreads();
      const int pos = 2 * t - (t & (stride - 1));
      const bool dir = ((pos & size) == 0);          // true => descending segment
      float a = ss[pos], b = ss[pos + stride];
      if ((a < b) == dir) {
        ss[pos] = b; ss[pos + stride] = a;
        int ia = si[pos]; si[pos] = si[pos + stride]; si[pos + stride] = ia;
      }
    }
  __syncthreads();
  cs[t] = ss[t]; cs[t + 1024] = ss[t + 1024];
  cidx[t] = si[t]; cidx[t + 1024] = si[t + 1024];
}

// ---------------------------------------------------------------- greedy NMS + mask + output
__global__ __launch_bounds__(256) void nms_kernel(const float* __restrict__ cs,
                                                  const int* __restrict__ cidx,
                                                  const float* __restrict__ locs,
                                                  const unsigned char* __restrict__ mask,
                                                  float* __restrict__ out) {
  __shared__ float sc[PRE_NMS_TOPK];
  __shared__ float lz[PRE_NMS_TOPK], ly[PRE_NMS_TOPK], lx[PRE_NMS_TOPK];
  __shared__ unsigned char sup[PRE_NMS_TOPK], kp[PRE_NMS_TOPK];
  const int tid = threadIdx.x;
  for (int j = tid; j < PRE_NMS_TOPK; j += 256) {
    sc[j] = cs[j];
    const int id = cidx[j];
    lz[j] = locs[id * 3 + 0]; ly[j] = locs[id * 3 + 1]; lx[j] = locs[id * 3 + 2];
    sup[j] = 0; kp[j] = 0;
  }
  for (int i = 0; i < PRE_NMS_TOPK; ++i) {
    __syncthreads();
    const bool keep_i = (!sup[i]) && (sc[i] >= 0.2f);
    if (tid == 0) kp[i] = keep_i ? 1 : 0;
    if (keep_i) {
      const float z0 = lz[i], y0 = ly[i], x0 = lx[i];
      for (int j = tid; j < PRE_NMS_TOPK; j += 256) {
        const float dz = lz[j] - z0, dy = ly[j] - y0, dx = lx[j] - x0;
        if (dz * dz + dy * dy + dx * dx < 64.f) sup[j] = 1;  // 8^2
      }
    }
  }
  __syncthreads();
  if (tid == 0) {
    int cnt = 0;
    for (int i = 0; i < PRE_NMS_TOPK && cnt < MAX_OUTPUT; ++i) {
      if (!kp[i]) continue;
      float z = lz[i], y = ly[i], x = lx[i], s = sc[i];
      int pz = (int)floorf(z), py = (int)floorf(y), px = (int)floorf(x);
      int cz = min(max(pz, 0), 63), cy = min(max(py, 0), 255), cx = min(max(px, 0), 255);
      bool in_range = (pz == cz) && (py == cy) && (px == cx);
      bool m = in_range && mask[((size_t)cz * 256 + cy) * 256 + cx] != 0;
      out[cnt * 4 + 0] = z; out[cnt * 4 + 1] = y; out[cnt * 4 + 2] = x;
      out[cnt * 4 + 3] = m ? s : 0.f;
      ++cnt;
    }
    for (; cnt < MAX_OUTPUT; ++cnt) {
      out[cnt * 4 + 0] = 0.f; out[cnt * 4 + 1] = 0.f;
      out[cnt * 4 + 2] = 0.f; out[cnt * 4 + 3] = 0.f;
    }
  }
}

// ---------------------------------------------------------------- launch
extern "C" void kernel_launch(void* const* d_in, const int* in_sizes, int n_in,
                              void* d_out, int out_size, void* d_ws, size_t ws_size,
                              hipStream_t stream) {
  const float* feature = (const float*)d_in[0];
  const unsigned char* mask = (const unsigned char*)d_in[1];
  const float* w0 = (const float*)d_in[2];
  const float* b0 = (const float*)d_in[3];
  const float* w1 = (const float*)d_in[4];
  const float* b1 = (const float*)d_in[5];
  const float* w_logit = (const float*)d_in[6];
  const float* b_logit = (const float*)d_in[7];
  const float* w_reg = (const float*)d_in[8];
  const float* b_reg = (const float*)d_in[9];
  float* out = (float*)d_out;

  char* ws = (char*)d_ws;
  const size_t SZ_ACT = (size_t)NVOX * CIN * 2;     // 32 MB f16
  const size_t SZ_W   = (size_t)27 * 256 * 256 * 2; // 3.4 MB f16
  _Float16* xf16 = (_Float16*)(ws);                 // also reused as y2
  _Float16* y1   = (_Float16*)(ws + SZ_ACT);
  _Float16* w0r  = (_Float16*)(ws + 2 * SZ_ACT);
  _Float16* w1r  = (_Float16*)(ws + 2 * SZ_ACT + SZ_W);
  float* scores  = (float*)(ws + 2 * SZ_ACT + 2 * SZ_W);
  float* locs    = (float*)(ws + 2 * SZ_ACT + 2 * SZ_W + (size_t)NVOX * 4);
  int*   hist    = (int*)(ws + 2 * SZ_ACT + 2 * SZ_W + (size_t)NVOX * 16);
  int*   ctr     = (int*)((char*)hist + 1024 * 4);
  float* cs      = (float*)((char*)ctr + 256);
  int*   cidx    = (int*)((char*)cs + PRE_NMS_TOPK * 4);

  // 1) precision conversion + weight swizzle
  cvt_f32_f16<<<(NVOX * CIN) / 256, 256, 0, stream>>>(feature, xf16, NVOX * CIN);
  reformat_w<<<(27 * 256 * 256) / 256, 256, 0, stream>>>(w0, w0r);
  reformat_w<<<(27 * 256 * 256) / 256, 256, 0, stream>>>(w1, w1r);

  // 2) two conv3d+gelu layers on the WMMA pipe (weights via TDM DMA)
  conv3_wmma<<<DD * HH, 256, 0, stream>>>(xf16, w0r, b0, y1);
  conv3_wmma<<<DD * HH, 256, 0, stream>>>(y1, w1r, b1, xf16);  // y2 reuses xf16

  // 3) detection head
  head_kernel<<<NVOX / 256, 256, 0, stream>>>(xf16, w_logit, b_logit, w_reg, b_reg,
                                              scores, locs);

  // 4) top-2048 radix select -> sort -> NMS -> masked output
  init_sel<<<8, 256, 0, stream>>>(hist, ctr, cs, cidx);
  hist_kernel<<<NVOX / 256, 256, 0, stream>>>(scores, hist);
  scan_kernel<<<1, 64, 0, stream>>>(hist, ctr);
  compact_kernel<<<NVOX / 256, 256, 0, stream>>>(scores, ctr, cs, cidx);
  sort_kernel<<<1, 1024, 0, stream>>>(cs, cidx);
  nms_kernel<<<1, 256, 0, stream>>>(cs, cidx, locs, mask, out);

  (void)in_sizes; (void)n_in; (void)out_size; (void)ws_size;
}
